// MyModel_50173807952598
// MI455X (gfx1250) — matrix operand order repaired and checked by reading, not compiled, not run on hardware
//
#include <hip/hip_runtime.h>
#include <hip/hip_bf16.h>
#include <stdint.h>

#define T_  512
#define B_  32
#define I_  512
#define H_  512
#define O_  512
#define G3H 1536        // 3*H
#define H2  1024        // 2*H
#define BH  (B_ * H_)   // 16384
#define NWG_SCAN 16     // workgroups per direction in the scan
#define SLICE    32     // H-columns owned per scan workgroup

#if defined(__has_builtin)
#  if __has_builtin(__builtin_amdgcn_global_load_async_to_lds_b128)
#    define HAVE_ASYNC_LDS 1
#  endif
#endif
#ifndef HAVE_ASYNC_LDS
#  define HAVE_ASYNC_LDS 0
#endif

typedef unsigned short ushort_t;
typedef __attribute__((ext_vector_type(16))) __bf16 v16bf;
typedef __attribute__((ext_vector_type(8)))  __bf16 v8bf;
typedef __attribute__((ext_vector_type(8)))  float  v8f;

#if HAVE_ASYNC_LDS
// exact pointee type from the builtin's prototype: GCC-vector int4
typedef int v4i_vs __attribute__((vector_size(16)));
typedef __attribute__((address_space(1))) v4i_vs* g_v4i_ptr;   // global
typedef __attribute__((address_space(3))) v4i_vs* l_v4i_ptr;   // LDS
#endif

__device__ __forceinline__ ushort_t f2bf(float f) {
  unsigned int u = __float_as_uint(f);
  unsigned int r = u + 0x7FFFu + ((u >> 16) & 1u);   // round-to-nearest-even
  return (ushort_t)(r >> 16);
}
__device__ __forceinline__ float bf2f(ushort_t h) {
  return __uint_as_float(((unsigned int)h) << 16);
}
__device__ __forceinline__ float sigmoidf_(float x) {
  return 1.0f / (1.0f + __expf(-x));
}

// ---- WMMA fragment loaders (bf16, 16x16x32) -------------------------------
// A (M=16 x K=32) row-major: lane<16 -> row m0+lane, K {k0..k0+7, k0+16..k0+23}
//                            lane>=16 -> row m0+lane-16, K {k0+8..15, k0+24..31}
__device__ __forceinline__ v16bf load_a_frag(const ushort_t* A, int lda,
                                             int m0, int k0, int lane) {
  int r  = lane & 15;
  int kk = k0 + ((lane >> 4) << 3);
  const ushort_t* p = A + (size_t)(m0 + r) * lda + kk;
  v8bf lo = *(const v8bf*)(p);
  v8bf hi = *(const v8bf*)(p + 16);
  return __builtin_shufflevector(lo, hi, 0,1,2,3,4,5,6,7,8,9,10,11,12,13,14,15);
}
// Same A-fragment pattern, but from a padded LDS tile (row stride 520 ush).
__device__ __forceinline__ v16bf load_a_frag_lds(const ushort_t hs[][520],
                                                 int m0, int k0, int lane) {
  int r  = lane & 15;
  int kk = k0 + ((lane >> 4) << 3);
  const ushort_t* p = &hs[m0 + r][kk];
  v8bf lo = *(const v8bf*)(p);
  v8bf hi = *(const v8bf*)(p + 16);
  return __builtin_shufflevector(lo, hi, 0,1,2,3,4,5,6,7,8,9,10,11,12,13,14,15);
}
// B (K=32 x N=16); weights stored (N rows x K cols) row-major:
// lane<16 -> col n0+lane, K {k0..k0+7, k0+16..k0+23}; lane>=16 -> K {k0+8..15, k0+24..31}
__device__ __forceinline__ v16bf load_b_frag(const ushort_t* Bw, int ldb,
                                             int n0, int k0, int lane) {
  int n  = n0 + (lane & 15);
  int kk = k0 + ((lane >> 4) << 3);
  const ushort_t* p = Bw + (size_t)n * ldb + kk;
  v8bf lo = *(const v8bf*)(p);
  v8bf hi = *(const v8bf*)(p + 16);
  return __builtin_shufflevector(lo, hi, 0,1,2,3,4,5,6,7,8,9,10,11,12,13,14,15);
}

// ---- f32 -> bf16 conversion kernel ----------------------------------------
__global__ void cvt_bf16_kernel(const float* __restrict__ src,
                                ushort_t* __restrict__ dst, int n) {
  for (int i = blockIdx.x * blockDim.x + threadIdx.x; i < n;
       i += gridDim.x * blockDim.x)
    dst[i] = f2bf(src[i]);
}

// ---- init hidden state + reset barrier counters ---------------------------
__global__ void init_h_kernel(const float* __restrict__ h_in, int layer,
                              float* __restrict__ hbuf,
                              ushort_t* __restrict__ hbbf,
                              unsigned int* __restrict__ bar) {
  int i = blockIdx.x * blockDim.x + threadIdx.x;
  if (i < 2) bar[i] = 0u;
  if (i < 2 * BH) {
    int dir = i / BH;
    int r   = i - dir * BH;
    float v = h_in[(size_t)(2 * layer + dir) * BH + r];
    hbuf[(size_t)dir * 2 * BH + r] = v;         // slot 0
    hbbf[(size_t)dir * 2 * BH + r] = f2bf(v);   // slot 0 bf16
  }
}

// ---- tiled bf16 WMMA GEMM:  C[M,N] = A[M,K] * B[N,K]^T + bias[N] ----------
// block tile 128x128; 8 waves laid out 4(M) x 2(N); wave tile 32x64
__global__ __launch_bounds__(256)
void gemm_bf16_kernel(const ushort_t* __restrict__ A, int lda,
                      const ushort_t* __restrict__ Bw, int ldb,
                      const float* __restrict__ bias,
                      ushort_t* __restrict__ Cbf, float* __restrict__ Cf,
                      int ldc, int K) {
  int w    = threadIdx.x >> 5;
  int lane = threadIdx.x & 31;
  int m0 = blockIdx.x * 128 + (w >> 1) * 32;
  int n0 = blockIdx.y * 128 + (w & 1) * 64;

  v8f acc[2][4] = {};
  for (int k0 = 0; k0 < K; k0 += 32) {
    v16bf a0 = load_a_frag(A, lda, m0,      k0, lane);
    v16bf a1 = load_a_frag(A, lda, m0 + 16, k0, lane);
    v16bf b0 = load_b_frag(Bw, ldb, n0,      k0, lane);
    v16bf b1 = load_b_frag(Bw, ldb, n0 + 16, k0, lane);
    v16bf b2 = load_b_frag(Bw, ldb, n0 + 32, k0, lane);
    v16bf b3 = load_b_frag(Bw, ldb, n0 + 48, k0, lane);
    acc[0][0] = __builtin_amdgcn_wmma_f32_16x16x32_bf16(false, a0, false, b0, (short)0, acc[0][0], false, false);
    acc[0][1] = __builtin_amdgcn_wmma_f32_16x16x32_bf16(false, a0, false, b1, (short)0, acc[0][1], false, false);
    acc[0][2] = __builtin_amdgcn_wmma_f32_16x16x32_bf16(false, a0, false, b2, (short)0, acc[0][2], false, false);
    acc[0][3] = __builtin_amdgcn_wmma_f32_16x16x32_bf16(false, a0, false, b3, (short)0, acc[0][3], false, false);
    acc[1][0] = __builtin_amdgcn_wmma_f32_16x16x32_bf16(false, a1, false, b0, (short)0, acc[1][0], false, false);
    acc[1][1] = __builtin_amdgcn_wmma_f32_16x16x32_bf16(false, a1, false, b1, (short)0, acc[1][1], false, false);
    acc[1][2] = __builtin_amdgcn_wmma_f32_16x16x32_bf16(false, a1, false, b2, (short)0, acc[1][2], false, false);
    acc[1][3] = __builtin_amdgcn_wmma_f32_16x16x32_bf16(false, a1, false, b3, (short)0, acc[1][3], false, false);
  }

  int g = lane >> 4, nn = lane & 15;
  #pragma unroll
  for (int mt = 0; mt < 2; ++mt) {
    #pragma unroll
    for (int nt = 0; nt < 4; ++nt) {
      int col = n0 + nt * 16 + nn;
      float bv = bias ? bias[col] : 0.0f;
      #pragma unroll
      for (int v = 0; v < 8; ++v) {
        int row = m0 + mt * 16 + v + 8 * g;
        float val = acc[mt][nt][v] + bv;
        if (Cf) Cf[(size_t)row * ldc + col] = val;
        else    Cbf[(size_t)row * ldc + col] = f2bf(val);
      }
    }
  }
}

// ---- persistent GRU scan: grid = (NWG_SCAN, 2 dirs), 384 threads ----------
// Each WG owns 32 H-columns (6 gate N-tiles x 2 M-tiles = 12 C-tiles, one per
// wave). Each wave keeps its entire w_hh panel (16 KB) register-resident for
// all 512 steps. h is async-staged to LDS once per step; gates staged through
// LDS; inter-WG sync via monotonic global atomic barrier + agent fences.
__global__ __launch_bounds__(384, 1)
void gru_scan_kernel(const ushort_t* __restrict__ gi_f,
                     const ushort_t* __restrict__ gi_b,
                     const ushort_t* __restrict__ whh_f,
                     const ushort_t* __restrict__ whh_b,
                     const float* __restrict__ bhh_f,
                     const float* __restrict__ bhh_b,
                     float* __restrict__ hbuf,      // [2 dir][2 slot][B*H] f32
                     ushort_t* __restrict__ hbbf,   // [2 dir][2 slot][B*H] bf16
                     ushort_t* __restrict__ ybase,  // (T,B,2H) bf16
                     float* __restrict__ hn_out,    // [2 dir][B*H] f32
                     unsigned int* __restrict__ bar)// [2 dir]
{
  const int dir = blockIdx.y;
  const ushort_t* gi  = dir ? gi_b  : gi_f;
  const ushort_t* whh = dir ? whh_b : whh_f;
  const float*    bhh = dir ? bhh_b : bhh_f;
  float*    hb  = hbuf + (size_t)dir * 2 * BH;
  ushort_t* hbb = hbbf + (size_t)dir * 2 * BH;
  unsigned int* mybar = bar + dir;

  const int js   = blockIdx.x * SLICE;
  const int w    = threadIdx.x >> 5;     // 0..11
  const int lane = threadIdx.x & 31;
  const int m0   = (w & 1) * 16;         // batch-row tile
  const int ntl  = w >> 1;               // 0..5 local gate N-tile
  const int gate = ntl >> 1;             // 0=r 1=z 2=n
  const int col  = gate * H_ + js + (ntl & 1) * 16;  // global w_hh row base
  const int g    = lane >> 4, nn = lane & 15;

  __shared__ ushort_t hshare[32][520];   // h bf16, padded rows (bank spread)
  __shared__ float    gh[32][100];       // 96 gate cols per slice (+pad)

  // ---- register-resident w_hh panel: 16 fragments = 128 VGPRs/lane ----
  v16bf breg[16];
  #pragma unroll
  for (int kt = 0; kt < 16; ++kt)
    breg[kt] = load_b_frag(whh, H_, col, kt * 32, lane);

  const int eb  = threadIdx.x >> 3;      // batch row   (valid for tid < 256)
  const int ej0 = (threadIdx.x & 7) * 4; // 4 slice-cols per thread

  for (int s = 0; s < T_; ++s) {
    const int t   = dir ? (T_ - 1 - s) : s;
    const int cur = s & 1, nxt = cur ^ 1;
    const ushort_t* hA = hbb + (size_t)cur * BH;

    // ---- stage h (32x512 bf16 = 32 KB) into LDS ----
#if HAVE_ASYNC_LDS
    for (int i = threadIdx.x; i < (B_ * H_) / 8; i += 384) {
      int r = i >> 6, c = (i & 63) * 8;
      __builtin_amdgcn_global_load_async_to_lds_b128(
          (g_v4i_ptr)(uintptr_t)(hA + (size_t)r * H_ + c),
          (l_v4i_ptr)(unsigned)(uintptr_t)&hshare[r][c],
          0, 0);
    }
#  if defined(__has_builtin) && __has_builtin(__builtin_amdgcn_s_wait_asynccnt)
    __builtin_amdgcn_s_wait_asynccnt(0);
#  else
    asm volatile("s_wait_asynccnt 0" ::: "memory");
#  endif
#else
    for (int i = threadIdx.x; i < (B_ * H_) / 8; i += 384) {
      int r = i >> 6, c = (i & 63) * 8;
      *(uint4*)&hshare[r][c] = *(const uint4*)(hA + (size_t)r * H_ + c);
    }
#endif
    __syncthreads();

    // ---- WMMA: this wave's 16x16 gate tile, B from registers ----
    v8f acc = {};
    #pragma unroll
    for (int kt = 0; kt < 16; ++kt) {
      v16bf a = load_a_frag_lds(hshare, m0, kt * 32, lane);
      acc = __builtin_amdgcn_wmma_f32_16x16x32_bf16(false, a, false, breg[kt], (short)0, acc, false, false);
    }
    #pragma unroll
    for (int v = 0; v < 8; ++v)
      gh[m0 + v + 8 * g][ntl * 16 + nn] = acc[v];
    __syncthreads();

    // ---- elementwise GRU gates (threads 0..255: 32 batch x 32 cols) ----
    if (threadIdx.x < 256) {
      const ushort_t* git = gi + ((size_t)t * B_ + eb) * G3H;
      const float*    hpr = hb + (size_t)cur * BH + (size_t)eb * H_;
      float*          hnw = hb + (size_t)nxt * BH + (size_t)eb * H_;
      ushort_t*       hnb = hbb + (size_t)nxt * BH + (size_t)eb * H_;
      ushort_t*       yw  = ybase + ((size_t)t * B_ + eb) * H2 + (size_t)dir * H_;
      #pragma unroll
      for (int u = 0; u < 4; ++u) {
        int jj = ej0 + u;
        int j  = js + jj;
        float ghr = gh[eb][jj]      + bhh[j];
        float ghz = gh[eb][32 + jj] + bhh[H_ + j];
        float ghn = gh[eb][64 + jj] + bhh[2 * H_ + j];
        float ir  = bf2f(git[j]);
        float iz  = bf2f(git[H_ + j]);
        float inn = bf2f(git[2 * H_ + j]);
        float hp  = hpr[j];
        float r   = sigmoidf_(ir + ghr);
        float z   = sigmoidf_(iz + ghz);
        float n   = tanhf(inn + r * ghn);
        float hv  = (1.0f - z) * n + z * hp;
        hnw[j] = hv;
        hnb[j] = f2bf(hv);
        yw[j]  = f2bf(hv);
        if (s == T_ - 1)
          hn_out[(size_t)dir * BH + (size_t)eb * H_ + j] = hv;
      }
    }

    // ---- inter-workgroup barrier (monotonic counter per direction) ----
    __threadfence();                      // release h writes to agent scope
    __syncthreads();
    if (threadIdx.x == 0) {
      __hip_atomic_fetch_add(mybar, 1u, __ATOMIC_RELEASE, __HIP_MEMORY_SCOPE_AGENT);
      unsigned int target = (unsigned int)(s + 1) * NWG_SCAN;
      while (__hip_atomic_load(mybar, __ATOMIC_ACQUIRE, __HIP_MEMORY_SCOPE_AGENT) < target)
        __builtin_amdgcn_s_sleep(1);
    }
    __syncthreads();
    __builtin_amdgcn_fence(__ATOMIC_ACQUIRE, "agent");  // invalidate stale WGP$
  }
}

// ---------------------------------------------------------------------------
extern "C" void kernel_launch(void* const* d_in, const int* in_sizes, int n_in,
                              void* d_out, int out_size, void* d_ws, size_t ws_size,
                              hipStream_t stream) {
  (void)in_sizes; (void)n_in; (void)out_size; (void)ws_size;

  const float* x     = (const float*)d_in[0];
  const float* h0    = (const float*)d_in[1];
  const float* wih0f = (const float*)d_in[2];
  const float* whh0f = (const float*)d_in[3];
  const float* bih0f = (const float*)d_in[4];
  const float* bhh0f = (const float*)d_in[5];
  const float* wih0b = (const float*)d_in[6];
  const float* whh0b = (const float*)d_in[7];
  const float* bih0b = (const float*)d_in[8];
  const float* bhh0b = (const float*)d_in[9];
  const float* wih1f = (const float*)d_in[10];
  const float* whh1f = (const float*)d_in[11];
  const float* bih1f = (const float*)d_in[12];
  const float* bhh1f = (const float*)d_in[13];
  const float* wih1b = (const float*)d_in[14];
  const float* whh1b = (const float*)d_in[15];
  const float* bih1b = (const float*)d_in[16];
  const float* bhh1b = (const float*)d_in[17];
  const float* fcw   = (const float*)d_in[18];
  const float* fcb   = (const float*)d_in[19];

  float* out_main = (float*)d_out;                    // (T,B,O) f32
  float* out_hn   = out_main + (size_t)T_ * B_ * O_;  // (4,B,H) f32

  char* p = (char*)d_ws;
  auto alloc = [&](size_t bytes) -> void* {
    void* r = (void*)p;
    p += (bytes + 255) & ~(size_t)255;
    return r;
  };

  ushort_t* xbf    = (ushort_t*)alloc(sizeof(ushort_t) * (size_t)T_ * B_ * I_);
  ushort_t* w0f_ih = (ushort_t*)alloc(sizeof(ushort_t) * (size_t)G3H * I_);
  ushort_t* w0f_hh = (ushort_t*)alloc(sizeof(ushort_t) * (size_t)G3H * H_);
  ushort_t* w0b_ih = (ushort_t*)alloc(sizeof(ushort_t) * (size_t)G3H * I_);
  ushort_t* w0b_hh = (ushort_t*)alloc(sizeof(ushort_t) * (size_t)G3H * H_);
  ushort_t* w1f_ih = (ushort_t*)alloc(sizeof(ushort_t) * (size_t)G3H * H2);
  ushort_t* w1f_hh = (ushort_t*)alloc(sizeof(ushort_t) * (size_t)G3H * H_);
  ushort_t* w1b_ih = (ushort_t*)alloc(sizeof(ushort_t) * (size_t)G3H * H2);
  ushort_t* w1b_hh = (ushort_t*)alloc(sizeof(ushort_t) * (size_t)G3H * H_);
  ushort_t* fcw_bf = (ushort_t*)alloc(sizeof(ushort_t) * (size_t)O_ * H2);
  ushort_t* gi_f   = (ushort_t*)alloc(sizeof(ushort_t) * (size_t)T_ * B_ * G3H);
  ushort_t* gi_b   = (ushort_t*)alloc(sizeof(ushort_t) * (size_t)T_ * B_ * G3H);
  ushort_t* y0     = (ushort_t*)alloc(sizeof(ushort_t) * (size_t)T_ * B_ * H2);
  ushort_t* y1     = (ushort_t*)alloc(sizeof(ushort_t) * (size_t)T_ * B_ * H2);
  float*    hbuf   = (float*)   alloc(sizeof(float)    * (size_t)2 * 2 * BH);
  ushort_t* hbbf   = (ushort_t*)alloc(sizeof(ushort_t) * (size_t)2 * 2 * BH);
  unsigned int* bar = (unsigned int*)alloc(sizeof(unsigned int) * 4);

  auto cvt = [&](const float* s, ushort_t* d, int n) {
    int blocks = (n + 255) / 256;
    if (blocks > 4096) blocks = 4096;
    cvt_bf16_kernel<<<blocks, 256, 0, stream>>>(s, d, n);
  };
  cvt(x,     xbf,    T_ * B_ * I_);
  cvt(wih0f, w0f_ih, G3H * I_);  cvt(whh0f, w0f_hh, G3H * H_);
  cvt(wih0b, w0b_ih, G3H * I_);  cvt(whh0b, w0b_hh, G3H * H_);
  cvt(wih1f, w1f_ih, G3H * H2);  cvt(whh1f, w1f_hh, G3H * H_);
  cvt(wih1b, w1b_ih, G3H * H2);  cvt(whh1b, w1b_hh, G3H * H_);
  cvt(fcw,   fcw_bf, O_ * H2);

  const dim3 blk(256);
  const dim3 blkscan(384);
  const dim3 ggi(T_ * B_ / 128, G3H / 128);   // (128, 12)
  const dim3 gfc(T_ * B_ / 128, O_ / 128);    // (128, 4)
  const dim3 gsc(NWG_SCAN, 2);

  // ---- layer 0 ----
  init_h_kernel<<<(2 * BH + 255) / 256, blk, 0, stream>>>(h0, 0, hbuf, hbbf, bar + 0);
  gemm_bf16_kernel<<<ggi, blk, 0, stream>>>(xbf, I_, w0f_ih, I_, bih0f, gi_f, nullptr, G3H, I_);
  gemm_bf16_kernel<<<ggi, blk, 0, stream>>>(xbf, I_, w0b_ih, I_, bih0b, gi_b, nullptr, G3H, I_);
  gru_scan_kernel<<<gsc, blkscan, 0, stream>>>(gi_f, gi_b, w0f_hh, w0b_hh, bhh0f, bhh0b,
                                               hbuf, hbbf, y0, out_hn + 0, bar + 0);
  // ---- layer 1 ----
  init_h_kernel<<<(2 * BH + 255) / 256, blk, 0, stream>>>(h0, 1, hbuf, hbbf, bar + 2);
  gemm_bf16_kernel<<<ggi, blk, 0, stream>>>(y0, H2, w1f_ih, H2, bih1f, gi_f, nullptr, G3H, H2);
  gemm_bf16_kernel<<<ggi, blk, 0, stream>>>(y0, H2, w1b_ih, H2, bih1b, gi_b, nullptr, G3H, H2);
  gru_scan_kernel<<<gsc, blkscan, 0, stream>>>(gi_f, gi_b, w1f_hh, w1b_hh, bhh1f, bhh1b,
                                               hbuf, hbbf, y1, out_hn + 2 * BH, bar + 2);
  // ---- final linear ----
  gemm_bf16_kernel<<<gfc, blk, 0, stream>>>(y1, H2, fcw_bf, H2, fcb, nullptr, out_main, O_, H2);
}
